// CausalAttention_64871186039338
// MI455X (gfx1250) — matrix operand order repaired and checked by reading, not compiled, not run on hardware
//
#include <hip/hip_runtime.h>
#include <hip/hip_bf16.h>

// ---------------------------------------------------------------------------
// Causal attention, B=8 S=2048 D=768, fp32 in/out.
// bf16 WMMA (v_wmma_f32_16x16x32_bf16) for all matmuls, fp32 flash softmax.
// All matmul loops are explicitly software-pipelined 1 iteration deep so the
// scheduler can keep loads in flight across WMMAs instead of draining
// loadcnt to 0 before every matrix op.
// ---------------------------------------------------------------------------

typedef __bf16 bf16;
typedef bf16  v4bf  __attribute__((ext_vector_type(4)));
typedef bf16  v8bf  __attribute__((ext_vector_type(8)));
typedef bf16  v16bf __attribute__((ext_vector_type(16)));
typedef float v8f   __attribute__((ext_vector_type(8)));

#define NB   8
#define SS   2048
#define DD   768
#define NROW (NB * SS)          // 16384

__device__ __forceinline__ v16bf cat8(v8bf lo, v8bf hi) {
    return __builtin_shufflevector(lo, hi, 0,1,2,3,4,5,6,7,8,9,10,11,12,13,14,15);
}

// A-matrix fragment (16x32 bf16, MxK). rowp points at (row m, k=0) of this
// K-chunk. Lanes 0-15: K={0..7,16..23}; lanes 16-31: K={8..15,24..31}.
__device__ __forceinline__ v16bf load_a16(const bf16* rowp, int half) {
    const v8bf lo = *(const v8bf*)(rowp + (half ? 8 : 0));
    const v8bf hi = *(const v8bf*)(rowp + 16 + (half ? 8 : 0));
    return cat8(lo, hi);
}

// B-matrix fragment (32x16 bf16, KxN). colp points at the start of column n's
// K-run (contiguous). Lanes 0-15: K={0..7,16..23}; lanes 16-31: K={8..15,24..31}.
__device__ __forceinline__ v16bf load_b16(const bf16* colp, int half) {
    const v8bf lo = *(const v8bf*)(colp + (half ? 8 : 0));
    const v8bf hi = *(const v8bf*)(colp + 16 + (half ? 8 : 0));
    return cat8(lo, hi);
}

__device__ __forceinline__ v8f wmma_bf16(v16bf a, v16bf b, v8f c) {
    return __builtin_amdgcn_wmma_f32_16x16x32_bf16(
        /*neg_a=*/false, a, /*neg_b=*/false, b,
        /*c_mod=*/(short)0, c, /*reuse_a=*/false, /*reuse_b=*/false);
}

// ---------------------------------------------------------------------------
// Kernel 1: fp32 -> bf16 elementwise convert (x), vectorized x4.
// ---------------------------------------------------------------------------
__global__ __launch_bounds__(256) void cvt_x(const float* __restrict__ x,
                                             bf16* __restrict__ xb, int n4) {
    int i = blockIdx.x * blockDim.x + threadIdx.x;
    if (i < n4) {
        const float4 v = ((const float4*)x)[i];
        v4bf o;
        o[0] = (bf16)v.x; o[1] = (bf16)v.y; o[2] = (bf16)v.z; o[3] = (bf16)v.w;
        ((v4bf*)xb)[i] = o;
    }
}

// ---------------------------------------------------------------------------
// Kernel 2: W[k][n] (fp32) -> Wt[n][k] (bf16), LDS tile transpose.
// ---------------------------------------------------------------------------
__global__ __launch_bounds__(256) void wtrans(const float* __restrict__ W,
                                              bf16* __restrict__ Wt) {
    __shared__ float t[16][17];
    const int k = blockIdx.y * 16 + threadIdx.y;
    const int n = blockIdx.x * 16 + threadIdx.x;
    t[threadIdx.y][threadIdx.x] = W[k * DD + n];
    __syncthreads();
    const int nn = blockIdx.x * 16 + threadIdx.y;
    const int kk = blockIdx.y * 16 + threadIdx.x;
    Wt[nn * DD + kk] = (bf16)t[threadIdx.x][threadIdx.y];
}

// ---------------------------------------------------------------------------
// Kernel 3: QKV projection GEMM. C[16384,768] = Xbf[16384,768] @ Wt^T.
// blockIdx.z: 0->Q, 1->K (row-major bf16), 2->V transposed: Vt[b][e][s].
// Block = 128 threads (4 waves); tile 64x128; wave computes 16x128.
// Software-pipelined: next A (d0+32) and next B (f+1) loads are issued
// before the current WMMA consumes its fragments.
// ---------------------------------------------------------------------------
__global__ __launch_bounds__(128) void qkv_gemm(
    const bf16* __restrict__ X,
    const bf16* __restrict__ WqT, const bf16* __restrict__ WkT,
    const bf16* __restrict__ WvT,
    bf16* __restrict__ Q, bf16* __restrict__ K, bf16* __restrict__ Vt) {
    const int z    = blockIdx.z;
    const bf16* Wt = (z == 0) ? WqT : (z == 1) ? WkT : WvT;
    const int wave = threadIdx.x >> 5;
    const int lane = threadIdx.x & 31;
    const int half = lane >> 4;
    const int ln   = lane & 15;
    const int m0   = blockIdx.x * 64 + wave * 16;
    const int n0   = blockIdx.y * 128;

    v8f acc[8];
    #pragma unroll
    for (int f = 0; f < 8; ++f) acc[f] = 0.0f;

    const bf16* arow = X + (size_t)(m0 + ln) * DD;
    const bf16* brow = Wt + (size_t)(n0 + ln) * DD;   // + f*16*DD per frag

    v16bf a = load_a16(arow, half);
    v16bf b = load_b16(brow, half);

    for (int d0 = 0; d0 < DD; d0 += 32) {
        // next A one K-step ahead
        const v16bf a_next =
            (d0 + 32 < DD) ? load_a16(arow + d0 + 32, half) : a;
        #pragma unroll
        for (int f = 0; f < 8; ++f) {
            v16bf b_next;
            if (f < 7) {
                b_next = load_b16(brow + (size_t)(f + 1) * 16 * DD + d0, half);
            } else if (d0 + 32 < DD) {
                b_next = load_b16(brow + d0 + 32, half);
            } else {
                b_next = b;
            }
            acc[f] = wmma_bf16(a, b, acc[f]);
            b = b_next;
        }
        a = a_next;
    }

    // C layout: lane l, VGPR i -> row M = i + 8*half, col N = l%16.
    if (z < 2) {
        bf16* O = z ? K : Q;
        #pragma unroll
        for (int f = 0; f < 8; ++f)
            #pragma unroll
            for (int i = 0; i < 8; ++i) {
                const int row = m0 + i + half * 8;
                const int col = n0 + f * 16 + ln;
                O[(size_t)row * DD + col] = (bf16)acc[f][i];
            }
    } else {
        #pragma unroll
        for (int f = 0; f < 8; ++f)
            #pragma unroll
            for (int i = 0; i < 8; ++i) {
                const int row = m0 + i + half * 8;     // global (b,s)
                const int bb  = row >> 11;             // /2048
                const int s   = row & (SS - 1);
                const int e   = n0 + f * 16 + ln;
                Vt[(size_t)bb * DD * SS + (size_t)e * SS + s] = (bf16)acc[f][i];
            }
    }
}

// ---------------------------------------------------------------------------
// Kernel 4: flash attention with causal mask.
// Block = 128 threads = 4 waves. wave -> (qg = wave>>1, eh = wave&1).
// Each wave: 16 query rows (q0 = bx*32 + qg*16), output columns
// [eh*384, eh*384+384): 24 f32 16x16 accumulators.
// 64-key KV tiles; scores via Q@K^T (K rows ARE B-columns: contiguous),
// PV via P@V with pre-transposed Vt (rows contiguous). Online softmax with
// shfl-xor reductions in 16-lane halves; P relayout C->A through wave-
// private padded LDS (no barriers; per-wave causal trip counts legal).
// Scores and PV loops are software-pipelined 1 deep; next K tile is
// prefetched (global_prefetch_b8) while this tile computes.
// ---------------------------------------------------------------------------
__global__ __launch_bounds__(128) void attn(
    const bf16* __restrict__ Q, const bf16* __restrict__ K,
    const bf16* __restrict__ Vt, float* __restrict__ Out) {
    __shared__ bf16 ps[4][16][72];   // 72-elem row pad -> conflict-free b128 reads

    const int wave = threadIdx.x >> 5;
    const int lane = threadIdx.x & 31;
    const int half = lane >> 4;
    const int ln   = lane & 15;
    const int qg   = wave >> 1;
    const int eh   = wave & 1;
    const int b    = blockIdx.y;
    const int q0   = blockIdx.x * 32 + qg * 16;
    const int e0   = eh * 384;

    const bf16* Qb = Q  + (size_t)b * SS * DD;
    const bf16* Kb = K  + (size_t)b * SS * DD;
    const bf16* Vb = Vt + (size_t)b * DD * SS;
    float*      Ob = Out + (size_t)b * SS * DD;

    const float scale = 0.03608439182435161f;   // 1/sqrt(768)
    const float L2E   = 1.44269504088896f;

    v8f acc[24];
    #pragma unroll
    for (int e = 0; e < 24; ++e) acc[e] = 0.0f;
    float mrun[8], lrun[8];
    #pragma unroll
    for (int i = 0; i < 8; ++i) { mrun[i] = -1e30f; lrun[i] = 0.0f; }

    const bf16* qrow = Qb + (size_t)(q0 + ln) * DD;
    const int ntiles = (q0 + 79) >> 6;          // 64-key tiles covering 0..q0+15

    for (int kt = 0; kt < ntiles; ++kt) {
        const int kv0 = kt * 64;
        const bf16* krow = Kb + (size_t)(kv0 + ln) * DD;  // + c*16*DD per frag

        // prefetch next K tile (2 x 32 row starts) into the cache hierarchy
        if (kt + 1 < ntiles) {
            __builtin_prefetch(Kb + (size_t)(kv0 + 64 + lane) * DD, 0, 1);
            __builtin_prefetch(Kb + (size_t)(kv0 + 96 + lane) * DD, 0, 1);
        }

        // ---- scores: S(16x64) = Qtile(16x768) @ K^T, pipelined 1 deep ----
        v8f s[4];
        #pragma unroll
        for (int c = 0; c < 4; ++c) s[c] = 0.0f;

        v16bf a = load_a16(qrow, half);
        v16bf bk[4];
        #pragma unroll
        for (int c = 0; c < 4; ++c)
            bk[c] = load_b16(krow + (size_t)c * 16 * DD, half);

        for (int d0 = 0; d0 < DD; d0 += 32) {
            const bool more = (d0 + 32 < DD);
            const v16bf a_next = more ? load_a16(qrow + d0 + 32, half) : a;
            v16bf bk_next[4];
            #pragma unroll
            for (int c = 0; c < 4; ++c)
                bk_next[c] = more
                    ? load_b16(krow + (size_t)c * 16 * DD + d0 + 32, half)
                    : bk[c];
            #pragma unroll
            for (int c = 0; c < 4; ++c)
                s[c] = wmma_bf16(a, bk[c], s[c]);
            a = a_next;
            #pragma unroll
            for (int c = 0; c < 4; ++c) bk[c] = bk_next[c];
        }

        // ---- online softmax (C layout: row M=i+8*half, col N=ln) ----
        float p[4][8];
        #pragma unroll
        for (int i = 0; i < 8; ++i) {
            const int qa = q0 + i + half * 8;
            float v[4];
            #pragma unroll
            for (int c = 0; c < 4; ++c) {
                v[c] = s[c][i] * scale;
                if (kv0 + c * 16 + ln > qa) v[c] = -1e30f;
            }
            float mx = fmaxf(fmaxf(v[0], v[1]), fmaxf(v[2], v[3]));
            mx = fmaxf(mx, __shfl_xor(mx, 1));
            mx = fmaxf(mx, __shfl_xor(mx, 2));
            mx = fmaxf(mx, __shfl_xor(mx, 4));
            mx = fmaxf(mx, __shfl_xor(mx, 8));
            const float mnew = fmaxf(mrun[i], mx);
            const float r    = __builtin_exp2f((mrun[i] - mnew) * L2E);
            mrun[i] = mnew;
            float rs = 0.0f;
            #pragma unroll
            for (int c = 0; c < 4; ++c) {
                p[c][i] = __builtin_exp2f((v[c] - mnew) * L2E);
                rs += p[c][i];
            }
            rs += __shfl_xor(rs, 1);
            rs += __shfl_xor(rs, 2);
            rs += __shfl_xor(rs, 4);
            rs += __shfl_xor(rs, 8);
            lrun[i] = lrun[i] * r + rs;
            #pragma unroll
            for (int e = 0; e < 24; ++e) acc[e][i] *= r;   // rescale output
        }

        // ---- stage P (bf16) C-layout -> LDS row-major (wave-private) ----
        #pragma unroll
        for (int i = 0; i < 8; ++i)
            #pragma unroll
            for (int c = 0; c < 4; ++c)
                ps[wave][i + half * 8][c * 16 + ln] = (bf16)p[c][i];

        // A-fragments of P from LDS (same-wave RAW: compiler inserts ds waits)
        const v16bf ap0 = cat8(*(const v8bf*)&ps[wave][ln][half * 8],
                               *(const v8bf*)&ps[wave][ln][16 + half * 8]);
        const v16bf ap1 = cat8(*(const v8bf*)&ps[wave][ln][32 + half * 8],
                               *(const v8bf*)&ps[wave][ln][48 + half * 8]);

        // ---- PV: acc += P(16x64) @ V(64 x 384-slice), pipelined 1 deep ----
        const bf16* vp0 = Vb + (size_t)(e0 + ln) * SS + kv0;   // + e*16*SS
        v16bf bv0 = load_b16(vp0, half);
        v16bf bv1 = load_b16(vp0 + 32, half);
        #pragma unroll
        for (int e = 0; e < 24; ++e) {
            v16bf bv0n = bv0, bv1n = bv1;
            if (e < 23) {
                const bf16* vpn = vp0 + (size_t)(e + 1) * 16 * SS;
                bv0n = load_b16(vpn, half);
                bv1n = load_b16(vpn + 32, half);
            }
            acc[e] = wmma_bf16(ap0, bv0, acc[e]);
            acc[e] = wmma_bf16(ap1, bv1, acc[e]);
            bv0 = bv0n;
            bv1 = bv1n;
        }
    }

    // ---- epilogue: normalize and store fp32 ----
    float inv[8];
    #pragma unroll
    for (int i = 0; i < 8; ++i) inv[i] = 1.0f / lrun[i];
    #pragma unroll
    for (int e = 0; e < 24; ++e)
        #pragma unroll
        for (int i = 0; i < 8; ++i) {
            const int row = q0 + i + half * 8;
            const int col = e0 + e * 16 + ln;
            Ob[(size_t)row * DD + col] = acc[e][i] * inv[i];
        }
}

// ---------------------------------------------------------------------------
// Launcher. Workspace layout (bytes):
//   [0)           Xbf : 25,165,824      [28,704,768)  Qbf : 25,165,824
//   [25,165,824)  WqT :  1,179,648      [53,870,592)  Kbf : 25,165,824
//   [26,345,472)  WkT :  1,179,648      [79,036,416)  Vt  : 25,165,824
//   [27,525,120)  WvT :  1,179,648      total ~104.2 MB
// ---------------------------------------------------------------------------
extern "C" void kernel_launch(void* const* d_in, const int* in_sizes, int n_in,
                              void* d_out, int out_size, void* d_ws, size_t ws_size,
                              hipStream_t stream) {
    const float* x  = (const float*)d_in[0];
    const float* wq = (const float*)d_in[1];
    const float* wk = (const float*)d_in[2];
    const float* wv = (const float*)d_in[3];
    float* out = (float*)d_out;

    char* ws = (char*)d_ws;
    bf16* Xbf = (bf16*)(ws);
    bf16* WqT = (bf16*)(ws + 25165824);
    bf16* WkT = (bf16*)(ws + 26345472);
    bf16* WvT = (bf16*)(ws + 27525120);
    bf16* Qb  = (bf16*)(ws + 28704768);
    bf16* Kb  = (bf16*)(ws + 53870592);
    bf16* Vt  = (bf16*)(ws + 79036416);

    // 1) convert x to bf16
    {
        const int n4 = NROW * DD / 4;               // 3,145,728
        cvt_x<<<dim3((n4 + 255) / 256), dim3(256), 0, stream>>>(x, Xbf, n4);
    }
    // 2) transpose weights to bf16 [out][in]
    {
        dim3 g(DD / 16, DD / 16), t(16, 16);
        wtrans<<<g, t, 0, stream>>>(wq, WqT);
        wtrans<<<g, t, 0, stream>>>(wk, WkT);
        wtrans<<<g, t, 0, stream>>>(wv, WvT);
    }
    // 3) QKV projections (z = 0:Q, 1:K, 2:V-transposed)
    {
        dim3 g(NROW / 64, DD / 128, 3), t(128);
        qkv_gemm<<<g, t, 0, stream>>>(Xbf, WqT, WkT, WvT, Qb, Kb, Vt);
    }
    // 4) causal flash attention
    {
        dim3 g(SS / 32, NB), t(128);
        attn<<<g, t, 0, stream>>>(Qb, Kb, Vt, out);
    }
}
